// ProteinFeatures_59992103191046
// MI455X (gfx1250) — matrix operand order repaired and checked by reading, not compiled, not run on hardware
//
#include <hip/hip_runtime.h>
#include <hip/hip_bf16.h>
#include <math.h>

typedef __attribute__((ext_vector_type(16))) _Float16 v16h;
typedef __attribute__((ext_vector_type(8)))  _Float16 v8h;
typedef __attribute__((ext_vector_type(8)))  float    v8f;

#define BB 8
#define LLEN 1024
#define KNN 30
#define NRBF 16
#define EDGE_F 128
#define EDGE_IN 416        // 16 pos + 25*16 rbf
#define FSTR 424           // padded LDS row stride (halves): 424*2 = 848 B (16B multiple)
#define NEDGE (BB*LLEN*KNN)

// ---------------- prep kernels ----------------

__global__ void cb_kernel(const float* __restrict__ X, float* __restrict__ Cb) {
  int idx = blockIdx.x * 256 + threadIdx.x;
  if (idx >= BB * LLEN) return;
  const float* p = X + (size_t)idx * 12;  // [4 atoms][3]
  float nx = p[0], ny = p[1], nz = p[2];
  float cax = p[3], cay = p[4], caz = p[5];
  float cx = p[6], cy = p[7], cz = p[8];
  float bx = cax - nx, by = cay - ny, bz = caz - nz;   // b = Ca - N
  float ex = cx - cax, ey = cy - cay, ez = cz - caz;   // c = C - Ca
  float ax = by * ez - bz * ey;                        // a = b x c
  float ay = bz * ex - bx * ez;
  float az = bx * ey - by * ex;
  Cb[idx*3+0] = -0.58273431f*ax + 0.56802827f*bx - 0.54067466f*ex + cax;
  Cb[idx*3+1] = -0.58273431f*ay + 0.56802827f*by - 0.54067466f*ey + cay;
  Cb[idx*3+2] = -0.58273431f*az + 0.56802827f*bz - 0.54067466f*ez + caz;
}

__global__ void w2h_kernel(const float* __restrict__ W, _Float16* __restrict__ Wh) {
  int idx = blockIdx.x * 256 + threadIdx.x;
  if (idx < EDGE_F * EDGE_IN) Wh[idx] = (_Float16)W[idx];
}

__global__ void zero_kernel(float* __restrict__ p, int n) {
  int idx = blockIdx.x * 256 + threadIdx.x;
  if (idx < n) p[idx] = 0.0f;
}

// ---------------- top-k kernel: one WG (8 waves) per (b,i) row ----------------

__global__ void topk_kernel(const float* __restrict__ X, const float* __restrict__ mask,
                            float* __restrict__ Dn_out, int* __restrict__ Eidx_out,
                            float* __restrict__ EidxF_out) {
  __shared__ float Dls[LLEN];
  __shared__ float rv[256];
  __shared__ int   ri[256];
  int bid = blockIdx.x;
  int b = bid / LLEN, i = bid % LLEN;
  int t = threadIdx.x;
  const float* Xb = X + (size_t)b * LLEN * 12;
  float cax = Xb[i*12+3], cay = Xb[i*12+4], caz = Xb[i*12+5];
  float mi = mask[b*LLEN + i];
  // pass 1: masked distances
  for (int j = t; j < LLEN; j += 256) {
    float dx = Xb[j*12+3]-cax, dy = Xb[j*12+4]-cay, dz = Xb[j*12+5]-caz;
    float m2 = mi * mask[b*LLEN + j];
    Dls[j] = m2 * sqrtf(dx*dx + dy*dy + dz*dz + 1e-6f);
  }
  __syncthreads();
  // row max for mask adjustment
  float mloc = -1e30f;
  for (int j = t; j < LLEN; j += 256) mloc = fmaxf(mloc, Dls[j]);
  rv[t] = mloc; __syncthreads();
  for (int s = 128; s > 0; s >>= 1) { if (t < s) rv[t] = fmaxf(rv[t], rv[t+s]); __syncthreads(); }
  float Dmax = rv[0];
  __syncthreads();
  // pass 2: D_adjust = D + (1 - mask2d) * Dmax
  for (int j = t; j < LLEN; j += 256) {
    float m2 = mi * mask[b*LLEN + j];
    Dls[j] += (1.0f - m2) * Dmax;
  }
  __syncthreads();
  // iterative argmin selection (stable: smallest index wins ties)
  for (int k = 0; k < KNN; ++k) {
    float bv = 3.3e38f; int bi = LLEN;
    for (int j = t; j < LLEN; j += 256) {
      float v = Dls[j];
      if (v < bv) { bv = v; bi = j; }
    }
    rv[t] = bv; ri[t] = bi; __syncthreads();
    for (int s = 128; s > 0; s >>= 1) {
      if (t < s) {
        float v2 = rv[t+s]; int i2 = ri[t+s];
        if (v2 < rv[t] || (v2 == rv[t] && i2 < ri[t])) { rv[t] = v2; ri[t] = i2; }
      }
      __syncthreads();
    }
    if (t == 0) {
      int widx = ri[0];
      size_t o = (size_t)(b*LLEN + i) * KNN + k;
      Eidx_out[o]  = widx;
      Dn_out[o]    = rv[0];
      EidxF_out[o] = (float)widx;
      Dls[widx] = 3.4e38f;
    }
    __syncthreads();
  }
}

// ---------------- fused feature + WMMA GEMM + LayerNorm kernel ----------------
// One workgroup (8 waves) per tile of 16 edges. Wave w computes output feature
// columns [16w, 16w+16) via 13x v_wmma_f32_16x16x32_f16 over K=416.

__global__ void edge_kernel(const float* __restrict__ X, const float* __restrict__ Cb,
                            const int* __restrict__ ridx, const int* __restrict__ chain,
                            const float* __restrict__ posW, const float* __restrict__ posb,
                            const _Float16* __restrict__ Wh,
                            const int* __restrict__ Eidx, const float* __restrict__ Dn,
                            const float* __restrict__ gamma, const float* __restrict__ beta,
                            float* __restrict__ Eout) {
  __shared__ __align__(32) _Float16 featLDS[16 * FSTR];
  __shared__ float aLDS[16][10][3];   // 5 atoms of residue i, 5 atoms of residue j
  __shared__ float Etile[16 * EDGE_F];
  __shared__ float muLDS[16], rsLDS[16];

  int t = threadIdx.x;
  int tileBase = blockIdx.x * 16;

  // phase 0: stage atom coords for the 16 edges (atom order N,Ca,C,O,Cb)
  if (t < 160) {
    int m = t / 10, am = t % 10;
    int n = tileBase + m;
    int b = n / (LLEN * KNN);
    int rem = n % (LLEN * KNN);
    int ii = rem / KNN;
    int j = Eidx[n];
    int res = (am < 5) ? ii : j;
    int a = am % 5;
    if (a < 4) {
      const float* p = X + (((size_t)b * LLEN + res) * 4 + a) * 3;
      aLDS[m][am][0] = p[0]; aLDS[m][am][1] = p[1]; aLDS[m][am][2] = p[2];
    } else {
      const float* p = Cb + ((size_t)b * LLEN + res) * 3;
      aLDS[m][am][0] = p[0]; aLDS[m][am][1] = p[1]; aLDS[m][am][2] = p[2];
    }
  }
  __syncthreads();

  // phase 1: build 16x416 fp16 feature tile. thread = (edge m, rbf channel r)
  {
    int m = t >> 4, r = t & 15;
    int n = tileBase + m;
    int b = n / (LLEN * KNN);
    int rem = n % (LLEN * KNN);
    int ii = rem / KNN;
    int j = Eidx[n];
    // positional feature: one-hot @ pos_W.T + pos_b  ==  pos_W[r][d] + pos_b[r]
    int off = ridx[b*LLEN + ii] - ridx[b*LLEN + j];
    int dch = (chain[b*LLEN + ii] == chain[b*LLEN + j]) ? 1 : 0;
    int dd = dch ? min(max(off + 32, 0), 64) : 65;
    featLDS[m*FSTR + r] = (_Float16)(posW[r*66 + dd] + posb[r]);
    // RBF params: mu = linspace(2,22,16), sigma = 20/16
    float mu = 2.0f + (20.0f / 15.0f) * (float)r;
    const float inv_sig = 1.0f / 1.25f;
    // block 0: Ca-Ca neighbor distance (adjusted, from top-k)
    float d0 = Dn[n];
    float z0 = (d0 - mu) * inv_sig;
    featLDS[m*FSTR + 16 + r] = (_Float16)__expf(-z0*z0);
    // 24 cross-atom pairs (atom enum: N=0,Ca=1,C=2,O=3,Cb=4)
    constexpr int PA[24] = {0,2,3,4,1,1,1,1,0,0,0,4,4,3,0,2,3,4,2,3,4,2,3,2};
    constexpr int PB[24] = {0,2,3,4,0,2,3,4,2,3,4,2,3,2,1,1,1,1,0,0,0,4,4,3};
    #pragma unroll
    for (int p = 0; p < 24; ++p) {
      float dx = aLDS[m][PA[p]][0] - aLDS[m][5 + PB[p]][0];
      float dy = aLDS[m][PA[p]][1] - aLDS[m][5 + PB[p]][1];
      float dz = aLDS[m][PA[p]][2] - aLDS[m][5 + PB[p]][2];
      float d = sqrtf(dx*dx + dy*dy + dz*dz + 1e-6f);
      float z = (d - mu) * inv_sig;
      featLDS[m*FSTR + 16 + (p + 1) * 16 + r] = (_Float16)__expf(-z*z);
    }
  }
  __syncthreads();

  // phase 2: WMMA GEMM. wave w owns feature cols [16w,16w+16); K=416 = 13*32
  {
    int lane = t & 31;
    int w = t >> 5;
    int mrow  = lane & 15;
    int koffA = (lane < 16) ? 0 : 16;      // A: lanes<16 hold K 0..15, lanes>=16 K 16..31 of each 32-chunk
    int fcol  = 16 * w + (lane & 15);
    int kbase = (lane < 16) ? 0 : 16;      // B: lanes<16 hold K 0..15, lanes>=16 K 16..31
    v8f c = {};
    const _Float16* arow = &featLDS[mrow * FSTR];
    const _Float16* brow = Wh + (size_t)fcol * EDGE_IN;
    #pragma unroll
    for (int kt = 0; kt < 13; ++kt) {
      v8h alo = *(const v8h*)(arow + 32*kt + koffA);
      v8h ahi = *(const v8h*)(arow + 32*kt + koffA + 8);
      v16h a;
      #pragma unroll
      for (int h = 0; h < 8; ++h) { a[h] = alo[h]; a[h + 8] = ahi[h]; }
      v16h bf = *(const v16h*)(brow + 32*kt + kbase);
      c = __builtin_amdgcn_wmma_f32_16x16x32_f16(false, a, false, bf,
                                                 (short)0, c, false, false);
    }
    // C layout: VGPR v -> row (lane<16 ? v : v+8), col = lane&15
    #pragma unroll
    for (int v = 0; v < 8; ++v) {
      int row = (lane < 16) ? v : v + 8;
      Etile[row * EDGE_F + fcol] = c[v];
    }
  }
  __syncthreads();

  // phase 3: LayerNorm over 128 features per edge
  if (t < 16) {
    float s = 0.0f, sq = 0.0f;
    for (int f = 0; f < EDGE_F; ++f) {
      float v = Etile[t * EDGE_F + f];
      s += v; sq += v * v;
    }
    float mu = s * (1.0f / EDGE_F);
    float var = sq * (1.0f / EDGE_F) - mu * mu;
    muLDS[t] = mu;
    rsLDS[t] = rsqrtf(var + 1e-5f);
  }
  __syncthreads();
  #pragma unroll
  for (int u = 0; u < 8; ++u) {
    int idx = t * 8 + u;               // 256*8 = 2048 = 16*128
    int row = idx >> 7, col = idx & 127;
    float v = (Etile[idx] - muLDS[row]) * rsLDS[row] * gamma[col] + beta[col];
    Eout[(size_t)(tileBase + row) * EDGE_F + col] = v;
  }
}

// ---------------- host launcher ----------------

extern "C" void kernel_launch(void* const* d_in, const int* in_sizes, int n_in,
                              void* d_out, int out_size, void* d_ws, size_t ws_size,
                              hipStream_t stream) {
  const float* X     = (const float*)d_in[0];
  const float* mask  = (const float*)d_in[1];
  const int*   ridx  = (const int*)d_in[2];
  const int*   chain = (const int*)d_in[3];
  const float* posW  = (const float*)d_in[4];
  const float* posb  = (const float*)d_in[5];
  const float* edgeW = (const float*)d_in[6];
  const float* gamma = (const float*)d_in[7];
  const float* beta  = (const float*)d_in[8];

  // workspace carve-up (all offsets 32B aligned)
  char* ws = (char*)d_ws;
  float*    Cb   = (float*)ws;                          // 8192*3*4      =  98304 B
  _Float16* Wh   = (_Float16*)(ws + 98304);             // 128*416*2     = 106496 B
  int*      Eidx = (int*)(ws + 98304 + 106496);         // 245760*4      = 983040 B
  float*    Dn   = (float*)(ws + 98304 + 106496 + 983040);

  float* out   = (float*)d_out;
  float* hV    = out;                                   // B*L*128 zeros
  float* Eout  = out + (size_t)BB * LLEN * EDGE_F;      // B*L*K*128
  float* EidxF = Eout + (size_t)NEDGE * EDGE_F;         // B*L*K (as float)

  zero_kernel<<<(BB*LLEN*EDGE_F + 255)/256, 256, 0, stream>>>(hV, BB*LLEN*EDGE_F);
  cb_kernel  <<<(BB*LLEN + 255)/256,        256, 0, stream>>>(X, Cb);
  w2h_kernel <<<(EDGE_F*EDGE_IN + 255)/256, 256, 0, stream>>>(edgeW, Wh);
  topk_kernel<<<BB*LLEN,                    256, 0, stream>>>(X, mask, Dn, Eidx, EidxF);
  edge_kernel<<<NEDGE/16,                   256, 0, stream>>>(X, Cb, ridx, chain, posW, posb,
                                                              Wh, Eidx, Dn, gamma, beta, Eout);
}
